// Bezier_27539330302394
// MI455X (gfx1250) — matrix operand order repaired and checked by reading, not compiled, not run on hardware
//
#include <hip/hip_runtime.h>
#include <hip/hip_bf16.h>

// ---------------------------------------------------------------------------
// Bezier Gaussian raster = (Ey @ Ex^T)/256, a 1024x1024x256 f32 GEMM.
// Strategy for MI455X (gfx1250, wave32):
//   Kernel 1: materialize Ex/Ey as f16 (1 MB total -> fully L2 resident).
//   Kernel 2: WMMA GEMM, v_wmma_f32_16x16x32_f16, one 16x16 tile per wave,
//             K=256 in 8 WMMA steps, f32 accumulate, f32 store.
// ---------------------------------------------------------------------------

typedef __attribute__((ext_vector_type(16))) _Float16 v16h;
typedef __attribute__((ext_vector_type(8)))  _Float16 v8h;
typedef __attribute__((ext_vector_type(8)))  float    v8f;

#define RES   1024
#define STEPS 256
// 1/(2*sigma^2) with sigma = 0.01
#define INV2S2 5000.0f

// ---------------------------------------------------------------------------
// Kernel 1: Ex[a][s] = exp(-(a/1024 - x[s])^2 * 5000)  (f16, row-major, ld=256)
//           Ey at offset RES*STEPS.  idx = which*2^18 + a*256 + s.
// ---------------------------------------------------------------------------
__global__ void bezier_exp_f16_kernel(const float* __restrict__ cp,
                                      _Float16* __restrict__ e_out) {
    int idx   = blockIdx.x * blockDim.x + threadIdx.x;   // 0 .. 2*RES*STEPS-1
    int which = idx >> 18;                               // 0 = Ex (x), 1 = Ey (y)
    int rem   = idx & ((RES * STEPS) - 1);
    int a     = rem >> 8;                                // pixel 0..1023
    int s     = rem & (STEPS - 1);                       // step  0..255

    // control_points is (3,2) row-major: [p0x p0y p1x p1y p2x p2y]
    float p0 = cp[0 + which];
    float p1 = cp[2 + which];
    float p2 = cp[4 + which];

    // lin_interp uses linspace(0,1,STEPS) (endpoint inclusive): tl = s/255
    // forward blend uses arange/STEPS (endpoint exclusive):     t  = s/256
    float tl = (float)s * (1.0f / (float)(STEPS - 1));
    float t  = (float)s * (1.0f / (float)STEPS);
    float av = p0 + (p1 - p0) * tl;
    float bv = p1 + (p2 - p1) * tl;
    float c  = av + t * (bv - av);                       // curve coordinate

    float g = (float)a * (1.0f / (float)RES);            // pixel coord / res
    float d = g - c;
    float e = __expf(-(d * d) * INV2S2);                 // v_exp_f32

    e_out[idx] = (_Float16)e;
}

// ---------------------------------------------------------------------------
// Kernel 2: out[i][j] = (1/256) * sum_s Ey[i][s] * Ex[j][s]
// One 16x16 output tile per wave; 8 waves per 256-thread block.
// A = Ey tile (16xK), B = Ex^T tile (Kx16). K stepped by 32 per WMMA.
// ---------------------------------------------------------------------------
__global__ void bezier_wmma_gemm_kernel(const _Float16* __restrict__ ews,
                                        float* __restrict__ out) {
    const _Float16* Ex = ews;                 // (1024, 256) f16 row-major
    const _Float16* Ey = ews + RES * STEPS;   // (1024, 256) f16 row-major

    int lane = threadIdx.x & 31;
    int wave = threadIdx.x >> 5;
    int tile = blockIdx.x * 8 + wave;         // 0 .. 4095
    int ti   = tile >> 6;                     // output row block (Ey rows)
    int tj   = tile & 63;                     // output col block (Ex rows)

    int  lm = lane & 15;
    bool hi = lane >= 16;

    // A-matrix f16 16x32 layout: low lanes K in [0,8)u[16,24), high lanes +8.
    // B-matrix f16 32x16 layout: low lanes K in [0,8)u[16,24), high lanes +8.
    const _Float16* arow = Ey + (ti * 16 + lm) * STEPS;
    const _Float16* brow = Ex + (tj * 16 + lm) * STEPS;
    int ka = hi ? 8 : 0;
    int kb = hi ? 8 : 0;

    v8f acc = {};

#pragma unroll
    for (int k0 = 0; k0 < STEPS; k0 += 32) {
        v8h a_lo = *(const v8h*)(arow + k0 + ka);        // global_load_b128
        v8h a_hi = *(const v8h*)(arow + k0 + ka + 16);
        v8h b_lo = *(const v8h*)(brow + k0 + kb);
        v8h b_hi = *(const v8h*)(brow + k0 + kb + 16);

        v16h A = __builtin_shufflevector(a_lo, a_hi,
                 0, 1, 2, 3, 4, 5, 6, 7, 8, 9, 10, 11, 12, 13, 14, 15);
        v16h B = __builtin_shufflevector(b_lo, b_hi,
                 0, 1, 2, 3, 4, 5, 6, 7, 8, 9, 10, 11, 12, 13, 14, 15);

        // D = A x B + C ; f32 accumulate. 8 args:
        // (neg_a, A, neg_b, B, c_mod, C, reuse_a, reuse_b)
        acc = __builtin_amdgcn_wmma_f32_16x16x32_f16(
            false, A, false, B, (short)0, acc, false, false);
    }

    // C/D layout: VGPR r -> row r (lanes 0-15) / row r+8 (lanes 16-31),
    // column = lane & 15.
    int col     = tj * 16 + lm;
    int rowbase = ti * 16 + (hi ? 8 : 0);
#pragma unroll
    for (int r = 0; r < 8; ++r) {
        out[(rowbase + r) * RES + col] = acc[r] * (1.0f / (float)STEPS);
    }
}

// ---------------------------------------------------------------------------
// d_in[0]: control_points (3,2) float32 (6 floats)
// d_out  : (1024,1024) float32
// d_ws   : needs 2*1024*256*2 bytes = 1 MB for f16 Ex/Ey
// ---------------------------------------------------------------------------
extern "C" void kernel_launch(void* const* d_in, const int* in_sizes, int n_in,
                              void* d_out, int out_size, void* d_ws, size_t ws_size,
                              hipStream_t stream) {
    const float* cp  = (const float*)d_in[0];
    _Float16*    ews = (_Float16*)d_ws;
    float*       out = (float*)d_out;

    // 2 * 1024 * 256 = 524288 elements, 256 threads/block -> 2048 blocks
    bezier_exp_f16_kernel<<<2048, 256, 0, stream>>>(cp, ews);

    // 4096 tiles of 16x16, 8 waves (256 threads) per block -> 512 blocks
    bezier_wmma_gemm_kernel<<<512, 256, 0, stream>>>(ews, out);
}
